// GNN_16801912062414
// MI455X (gfx1250) — matrix operand order repaired and checked by reading, not compiled, not run on hardware
//
#include <hip/hip_runtime.h>
#include <hip/hip_bf16.h>

typedef __attribute__((ext_vector_type(16))) __bf16 v16bf;
typedef __attribute__((ext_vector_type(8)))  float  v8f;

// ---------------------------------------------------------------- utilities
__global__ void k_fill(float* __restrict__ p, float v, int n) {
  int i = blockIdx.x * blockDim.x + threadIdx.x;
  if (i < n) p[i] = v;
}

// deg[col] += 1 per edge (deg pre-initialized to 1.0 for the self-loop)
__global__ void k_deg(const long long* __restrict__ col, float* __restrict__ deg, int E) {
  int e = blockIdx.x * blockDim.x + threadIdx.x;
  if (e < E) atomicAdd(&deg[(int)col[e]], 1.0f);
}

// deg -> dinv in place (deg >= 1 always, so no zero guard needed)
__global__ void k_dinv(float* __restrict__ d, int n) {
  int i = blockIdx.x * blockDim.x + threadIdx.x;
  if (i < n) d[i] = rsqrtf(d[i]);
}

// ---------------------------------------------------------------- WMMA GEMM
// T[N,NF] = (RELU ? relu(A) : A)[N,KF] @ W[KF,NF], bf16 inputs, f32 accum.
// One wave = one 16x16 output tile. ISA 7.12.2 16-bit layouts:
//   A (16xK): lane m=L%16 is row; elems 0-7 -> K {0-7 | 8-15}, elems 8-15 -> K {16-23 | 24-31}
//   B (Kx16): lane m=L%16 is col; elems 0-7 -> K {0-7 | 8-15}, elems 8-15 -> K {16-23 | 24-31}
//   C/D f32 : vgpr r -> row r + 8*(L>=16), col L%16
// Loads are branchless: addresses clamped in-range, zero-padding via select.
template<int KF, int NF, bool RELU>
__global__ __launch_bounds__(256) void k_gemm(const float* __restrict__ A,
                                              const float* __restrict__ W,
                                              float* __restrict__ T, int n) {
  const int lane = threadIdx.x & 31;
  const int m    = lane & 15;
  const int hi   = lane >> 4;
  constexpr int CT = (NF + 15) / 16;            // column tiles
  const int gw    = blockIdx.x * (blockDim.x >> 5) + (threadIdx.x >> 5);
  const int tiles = ((n + 15) >> 4) * CT;
  if (gw >= tiles) return;                      // wave-uniform: EXEC stays full
  const int nt   = gw / CT;
  const int ct   = gw % CT;
  const int arow = nt * 16 + m;
  const int arowc = (arow < n) ? arow : (n - 1);   // clamped, always in-range
  const int bcol = ct * 16 + m;                    // NF % 16 == 0 here

  v8f c = {};
  constexpr int KSTEPS = (KF + 31) / 32;
  #pragma unroll
  for (int ks = 0; ks < KSTEPS; ++ks) {
    v16bf a, b;
    #pragma unroll
    for (int e = 0; e < 16; ++e) {
      const int ka  = ks * 32 + e + ((e >= 8) ? 8 : 0) + hi * 8;
      const int kac = (ka < KF) ? ka : 0;          // folds away for KF=32/64
      float av = A[arowc * KF + kac];
      if (RELU) av = fmaxf(av, 0.0f);
      av = (ka < KF) ? av : 0.0f;                  // v_cndmask, no branch
      a[e] = (__bf16)av;
      const int kb  = ks * 32 + e + ((e >= 8) ? 8 : 0) + hi * 8;
      const int kbc = (kb < KF) ? kb : 0;
      float bv = W[kbc * NF + bcol];
      bv = (kb < KF) ? bv : 0.0f;
      b[e] = (__bf16)bv;
    }
    c = __builtin_amdgcn_wmma_f32_16x16x32_bf16(false, a, false, b,
                                                (short)0, c, false, false);
  }
  #pragma unroll
  for (int r = 0; r < 8; ++r) {
    const int orow = nt * 16 + r + hi * 8;
    if (orow < n) T[orow * NF + bcol] = c[r];
  }
}

// ------------------------------------------------- self-loop + bias init
// h[i,f] = b[f] + t[i,f] * dinv[i]^2   (pre-relu accumulator init)
template<int F>
__global__ void k_selfbias(const float* __restrict__ t, const float* __restrict__ dinv,
                           const float* __restrict__ b, float* __restrict__ h, int n) {
  int i = blockIdx.x * blockDim.x + threadIdx.x;
  if (i < n * F) {
    int node = i / F, f = i % F;
    float d = dinv[node];
    h[i] = b[f] + t[i] * d * d;
  }
}

// ------------------------------------------------- edge gather/scatter-add
// One wave per edge: h[col] += t[row] * dinv[row]*dinv[col]
template<int F>
__global__ __launch_bounds__(256) void k_edge(const long long* __restrict__ row,
                                              const long long* __restrict__ col,
                                              const float* __restrict__ dinv,
                                              const float* __restrict__ t,
                                              float* __restrict__ h, int E) {
  const int tid  = blockIdx.x * blockDim.x + threadIdx.x;
  const int e    = tid >> 5;
  const int lane = tid & 31;
  if (e >= E) return;
  const int r = (int)row[e];
  const int c = (int)col[e];
  const float nrm = dinv[r] * dinv[c];
  if (F == 64) {
    const float2 v = *(const float2*)(t + r * 64 + lane * 2);
    atomicAdd(&h[c * 64 + lane * 2    ], v.x * nrm);
    atomicAdd(&h[c * 64 + lane * 2 + 1], v.y * nrm);
  } else {
    const float v = t[r * 32 + lane];
    atomicAdd(&h[c * 32 + lane], v * nrm);
  }
}

// ------------------------------------------------- per-graph node counts
__global__ void k_count(const long long* __restrict__ batch, float* __restrict__ cnt, int n) {
  int i = blockIdx.x * blockDim.x + threadIdx.x;
  if (i < n) atomicAdd(&cnt[(int)batch[i]], 1.0f);
}

// ------------------------------------------------- head GEMM fused with pooling
// sums[g, j] += relu(h2)[node, :] @ Wl[:, j] + bl[j], via per-block LDS scoreboard.
// KF = 32 (one WMMA K-step), NF padded from 5 to 16, cols >= 5 discarded.
__global__ __launch_bounds__(256) void k_gemm3_pool(const float* __restrict__ H,
                                                    const float* __restrict__ Wl,
                                                    const float* __restrict__ bl,
                                                    const long long* __restrict__ batch,
                                                    float* __restrict__ sums, int n) {
  __shared__ float ls[64 * 5];
  for (int i = threadIdx.x; i < 320; i += blockDim.x) ls[i] = 0.0f;
  __syncthreads();

  const int lane = threadIdx.x & 31;
  const int m    = lane & 15;
  const int hi   = lane >> 4;
  const int gw    = blockIdx.x * (blockDim.x >> 5) + (threadIdx.x >> 5);
  const int tiles = (n + 15) >> 4;

  if (gw < tiles) {                              // wave-uniform predicate
    const int arow  = gw * 16 + m;
    const int arowc = (arow < n) ? arow : (n - 1);
    const int mc    = (m < 5) ? m : 0;
    v16bf a, b;
    #pragma unroll
    for (int e = 0; e < 16; ++e) {
      const int ka = e + ((e >= 8) ? 8 : 0) + hi * 8;      // 0..31, always < 32
      a[e] = (__bf16)fmaxf(H[arowc * 32 + ka], 0.0f);
      const int kb = e + ((e >= 8) ? 8 : 0) + hi * 8;      // 0..31
      float bv = Wl[kb * 5 + mc];
      b[e] = (__bf16)((m < 5) ? bv : 0.0f);
    }
    v8f c = {};
    c = __builtin_amdgcn_wmma_f32_16x16x32_bf16(false, a, false, b,
                                                (short)0, c, false, false);
    #pragma unroll
    for (int r = 0; r < 8; ++r) {
      const int orow = gw * 16 + r + hi * 8;
      if (m < 5 && orow < n) {
        const int g = (int)batch[orow];
        atomicAdd(&ls[g * 5 + m], c[r] + bl[m]);           // DS atomic
      }
    }
  }
  __syncthreads();
  for (int i = threadIdx.x; i < 320; i += blockDim.x) {
    const float v = ls[i];
    if (v != 0.0f) atomicAdd(&sums[i], v);
  }
}

// ------------------------------------------------- final mean
__global__ void k_final(const float* __restrict__ sums, const float* __restrict__ cnt,
                        float* __restrict__ out) {
  int i = threadIdx.x;
  if (i < 320) out[i] = sums[i] / fmaxf(cnt[i / 5], 1.0f);
}

// ================================================================ launcher
extern "C" void kernel_launch(void* const* d_in, const int* in_sizes, int n_in,
                              void* d_out, int out_size, void* d_ws, size_t ws_size,
                              hipStream_t stream) {
  (void)n_in; (void)out_size; (void)ws_size;
  const float*     x   = (const float*)d_in[0];      // [N,7]
  const long long* ei  = (const long long*)d_in[1];  // [2,E] int64
  const long long* bt  = (const long long*)d_in[2];  // [N]   int64
  const float* W1 = (const float*)d_in[3];           // [7,64]
  const float* b1 = (const float*)d_in[4];           // [64]
  const float* W2 = (const float*)d_in[5];           // [64,32]
  const float* b2 = (const float*)d_in[6];           // [32]
  const float* Wl = (const float*)d_in[7];           // [32,5]
  const float* bl = (const float*)d_in[8];           // [5]
  float* out = (float*)d_out;                        // [64,5]

  const int N = in_sizes[0] / 7;
  const int E = in_sizes[1] / 2;
  const long long* row = ei;
  const long long* col = ei + E;

  // workspace layout (256B aligned): dinv | t | h | sums(320)+cnt(64)
  char* ws = (char*)d_ws;
  auto al = [](size_t v) { return (v + 255) & ~(size_t)255; };
  size_t o = 0;
  float* dinv = (float*)(ws + o); o = al(o + (size_t)N * 4);
  float* t    = (float*)(ws + o); o = al(o + (size_t)N * 64 * 4);
  float* h    = (float*)(ws + o); o = al(o + (size_t)N * 64 * 4);
  float* sums = (float*)(ws + o); o += 320 * 4;
  float* cnt  = (float*)(ws + o);                 // contiguous after sums

  const int TB = 256;
  auto cdiv = [](long long a, long long b) { return (int)((a + b - 1) / b); };

  // degrees / normalization
  k_fill<<<cdiv(N, TB), TB, 0, stream>>>(dinv, 1.0f, N);     // self-loop
  k_fill<<<1, 512, 0, stream>>>(sums, 0.0f, 384);            // sums + cnt
  k_deg <<<cdiv(E, TB), TB, 0, stream>>>(col, dinv, E);
  k_dinv<<<cdiv(N, TB), TB, 0, stream>>>(dinv, N);

  // layer 1: t1 = x@W1 ; h1 = b1 + self + scatter edges  (pre-relu)
  k_gemm<7, 64, false><<<cdiv((long long)((N + 15) / 16) * 4, 8), TB, 0, stream>>>(x, W1, t, N);
  k_selfbias<64><<<cdiv((long long)N * 64, TB), TB, 0, stream>>>(t, dinv, b1, h, N);
  k_edge<64><<<cdiv((long long)E * 32, TB), TB, 0, stream>>>(row, col, dinv, t, h, E);

  // layer 2: t2 = relu(h1)@W2 ; h2 = b2 + self + scatter edges  (pre-relu)
  k_gemm<64, 32, true><<<cdiv((long long)((N + 15) / 16) * 2, 8), TB, 0, stream>>>(h, W2, t, N);
  k_selfbias<32><<<cdiv((long long)N * 32, TB), TB, 0, stream>>>(t, dinv, b2, h, N);
  k_edge<32><<<cdiv((long long)E * 32, TB), TB, 0, stream>>>(row, col, dinv, t, h, E);

  // head + pooling
  k_count<<<cdiv(N, TB), TB, 0, stream>>>(bt, cnt, N);
  k_gemm3_pool<<<cdiv((long long)((N + 15) / 16), 8), TB, 0, stream>>>(h, Wl, bl, bt, sums, N);
  k_final<<<1, 512, 0, stream>>>(sums, cnt, out);
}
